// SplitCrossEntropyLoss_3882650435985
// MI455X (gfx1250) — matrix operand order repaired and checked by reading, not compiled, not run on hardware
//
#include <hip/hip_runtime.h>
#include <hip/hip_bf16.h>
#include <stdint.h>

// ---------------------------------------------------------------------------
// Adaptive softmax (split cross-entropy) for MI455X / gfx1250.
// GEMM via v_wmma_f32_16x16x32_bf16 (wave32) with ASYNC global->LDS staging
// (global_load_async_to_lds_b128 + s_wait_asynccnt), online LSE partials,
// exact fp32 target logits, tiny final reductions.
// ---------------------------------------------------------------------------

typedef __attribute__((ext_vector_type(16))) __bf16 v16bf;
typedef __attribute__((ext_vector_type(8)))  float  v8f;

struct Q2 { uint4 x, y; };   // 32 bytes -> bit_cast to v16bf

#define HDIM   1024
#define NROWS  4096
#define NC     50176   // padded combined columns: [0,4000)=head W, 4000..4001 tail vecs,
                       // 4002..4095 pad, 4096..20095 tail1(16000), 20096..50095 tail2(30000), pad to 50176
#define NTILES 392     // NC / 128
#define MT     128
#define NTB    128
#define KB     32
#define LDSS   48      // LDS row stride (uint16 units), multiple of 8 -> 16B-aligned b128 ds loads
#define NEGBIG (-3.0e38f)

__device__ __forceinline__ unsigned short f2bf(float f) {
  unsigned int u = __float_as_uint(f);
  u += 0x7FFFu + ((u >> 16) & 1u);       // round-to-nearest-even
  return (unsigned short)(u >> 16);
}

// Generic pointers into LDS carry the wave-relative LDS byte offset in the low
// 32 bits (aperture mapping).
__device__ __forceinline__ unsigned lds_off(const void* p) {
  return (unsigned)(unsigned long long)p;
}

// CDNA5 async copy: memory -> LDS, no VGPR data, tracked with ASYNCcnt.
__device__ __forceinline__ void async_g2lds_b128(unsigned loff, const void* gaddr) {
  asm volatile("global_load_async_to_lds_b128 %0, %1, off"
               :: "v"(loff), "v"(gaddr)
               : "memory");
}

__device__ __forceinline__ void wait_asynccnt0() {
  asm volatile("s_wait_asynccnt 0x0" ::: "memory");
}

// ---------------------------------------------------------------------------
// Kernel 1: build padded bf16 combined weight (+fp32 bias array) and bf16 hiddens.
// grid.x = NC + NROWS rows, 128 threads/row, 8 elements/thread.
// ---------------------------------------------------------------------------
__global__ void __launch_bounds__(128) k_convert(
    const float* __restrict__ weight, const float* __restrict__ bias,
    const float* __restrict__ hiddens, const float* __restrict__ tailv,
    const float* __restrict__ tailb,
    unsigned short* __restrict__ Wb, float* __restrict__ biasb,
    unsigned short* __restrict__ Hb)
{
  const int j  = blockIdx.x;
  const int t  = threadIdx.x;
  const int k0 = t * 8;

  const float* src;
  unsigned short* dst;
  if (j < NC) {
    float b = -1.0e30f;                    // pad columns get -inf-ish bias
    src = nullptr;
    if (j < 4000)       { src = weight + (size_t)j * HDIM;          b = bias[j]; }
    else if (j < 4002)  { src = tailv  + (size_t)(j - 4000) * HDIM; b = tailb[j - 4000]; }
    else if (j < 4096)  { /* pad */ }
    else if (j < 50096) { src = weight + (size_t)(j - 96) * HDIM;   b = bias[j - 96]; }
    dst = Wb + (size_t)j * HDIM;
    if (t == 0) biasb[j] = b;
    if (!src) {
      uint4 z = {0u, 0u, 0u, 0u};
      *(uint4*)(dst + k0) = z;
      return;
    }
  } else {
    const int r = j - NC;
    src = hiddens + (size_t)r * HDIM;
    dst = Hb + (size_t)r * HDIM;
  }
  const float4* s4 = (const float4*)(src + k0);
  float4 f0 = s4[0], f1 = s4[1];
  uint4 pk;
  pk.x = (unsigned)f2bf(f0.x) | ((unsigned)f2bf(f0.y) << 16);
  pk.y = (unsigned)f2bf(f0.z) | ((unsigned)f2bf(f0.w) << 16);
  pk.z = (unsigned)f2bf(f1.x) | ((unsigned)f2bf(f1.y) << 16);
  pk.w = (unsigned)f2bf(f1.z) | ((unsigned)f2bf(f1.w) << 16);
  *(uint4*)(dst + k0) = pk;
}

// ---------------------------------------------------------------------------
// Kernel 2: tiled bf16 WMMA GEMM with fused online softmax partials.
// Block = 256 threads (8 waves, 2x4 wave grid). Tile 128(M) x 128(N) x 1024(K).
// Staging: global_load_async_to_lds_b128 double-buffered; s_wait_asynccnt only
// at the end of each iteration so the copy overlaps the 8 WMMAs.
// ---------------------------------------------------------------------------
__global__ void __launch_bounds__(256) k_gemm_lse(
    const unsigned short* __restrict__ Wb,
    const unsigned short* __restrict__ Hb,
    const float* __restrict__ biasb,
    float2* __restrict__ partial)
{
  __shared__ unsigned short As[2][MT * LDSS];
  __shared__ unsigned short Bs[2][NTB * LDSS];
  __shared__ float2 SStat[4][MT];

  const int nt   = blockIdx.x;
  const int mt   = blockIdx.y;
  const int tid  = threadIdx.x;
  const int lane = tid & 31;
  const int wave = tid >> 5;
  const int wm   = wave >> 2;    // 0..1  (rows  wm*64 .. +64)
  const int wn   = wave & 3;     // 0..3  (cols  wn*32 .. +32)
  const int l15  = lane & 15;
  const bool hi  = lane >= 16;

  // Per-thread staging slots (hand-unrolled; 2 x 16B for A, 2 x 16B for B).
  const int iA0 = tid, iA1 = tid + 256;          // of 512 b128 transfers for A
  const int mA0 = iA0 >> 2, kgA0 = (iA0 & 3) << 3;
  const int mA1 = iA1 >> 2, kgA1 = (iA1 & 3) << 3;
  const int nB0 = iA0 >> 2, kgB0 = kgA0;
  const int nB1 = iA1 >> 2, kgB1 = kgA1;

  const unsigned short* gA0 = Hb + (size_t)(mt * MT + mA0) * HDIM + kgA0;
  const unsigned short* gA1 = Hb + (size_t)(mt * MT + mA1) * HDIM + kgA1;
  const unsigned short* gB0 = Wb + (size_t)(nt * NTB + nB0) * HDIM + kgB0;
  const unsigned short* gB1 = Wb + (size_t)(nt * NTB + nB1) * HDIM + kgB1;

  const unsigned lA0[2] = { lds_off(&As[0][mA0 * LDSS + kgA0]),
                            lds_off(&As[1][mA0 * LDSS + kgA0]) };
  const unsigned lA1[2] = { lds_off(&As[0][mA1 * LDSS + kgA1]),
                            lds_off(&As[1][mA1 * LDSS + kgA1]) };
  const unsigned lB0[2] = { lds_off(&Bs[0][nB0 * LDSS + kgB0]),
                            lds_off(&Bs[1][nB0 * LDSS + kgB0]) };
  const unsigned lB1[2] = { lds_off(&Bs[0][nB1 * LDSS + kgB1]),
                            lds_off(&Bs[1][nB1 * LDSS + kgB1]) };

  auto stage = [&](int buf, int k0) {
    async_g2lds_b128(lA0[buf], gA0 + k0);
    async_g2lds_b128(lA1[buf], gA1 + k0);
    async_g2lds_b128(lB0[buf], gB0 + k0);
    async_g2lds_b128(lB1[buf], gB1 + k0);
  };

  const v8f vzero = {0.f, 0.f, 0.f, 0.f, 0.f, 0.f, 0.f, 0.f};
  v8f acc[4][2];
#pragma unroll
  for (int mi = 0; mi < 4; mi++)
#pragma unroll
    for (int ni = 0; ni < 2; ni++) acc[mi][ni] = vzero;

  stage(0, 0);
  wait_asynccnt0();
  __syncthreads();

  const int NKS = HDIM / KB;  // 32
  for (int ks = 0; ks < NKS; ks++) {
    const int buf = ks & 1;
    const bool more = (ks + 1 < NKS);
    if (more) stage(buf ^ 1, (ks + 1) * KB);   // async: overlaps the WMMAs below

    // B fragments: 32x16 bf16; lanes 0-15 hold K 0-7 & 16-23, lanes 16-31 K 8-15 & 24-31,
    // column n = lane&15, contiguous K pairs along VGPRs.
    v16bf bfrag[2];
#pragma unroll
    for (int ni = 0; ni < 2; ni++) {
      const int n_l = wn * 32 + ni * 16 + l15;
      const int kb  = hi ? 8 : 0;
      Q2 q;
      q.x = *(const uint4*)&Bs[buf][n_l * LDSS + kb];
      q.y = *(const uint4*)&Bs[buf][n_l * LDSS + kb + 16];
      bfrag[ni] = __builtin_bit_cast(v16bf, q);
    }

#pragma unroll
    for (int mi = 0; mi < 4; mi++) {
      // A fragment: 16x32 bf16; lanes 0-15: K 0-7 & 16-23, lanes 16-31: K 8-15 & 24-31.
      const int m_l = wm * 64 + mi * 16 + l15;
      const int ka  = hi ? 8 : 0;
      Q2 q;
      q.x = *(const uint4*)&As[buf][m_l * LDSS + ka];
      q.y = *(const uint4*)&As[buf][m_l * LDSS + ka + 16];
      const v16bf afrag = __builtin_bit_cast(v16bf, q);
#pragma unroll
      for (int ni = 0; ni < 2; ni++) {
        acc[mi][ni] = __builtin_amdgcn_wmma_f32_16x16x32_bf16(
            false, afrag, false, bfrag[ni], (short)0, acc[mi][ni], false, false);
      }
    }

    if (more) wait_asynccnt0();  // next tile resident before the barrier
    __syncthreads();
  }

  // ---- epilogue: bias + per-row (max, sum_exp) over this wave's 32 columns ----
  const int   colbase = nt * NTB + wn * 32;
  const float b0 = biasb[colbase + l15];
  const float b1 = biasb[colbase + 16 + l15];

#pragma unroll
  for (int mi = 0; mi < 4; mi++) {
#pragma unroll
    for (int r = 0; r < 8; r++) {
      // C layout: VGPR r -> row (lanes<16: r, lanes>=16: r+8), col = lane&15.
      float a0 = acc[mi][0][r] + b0;
      float a1 = acc[mi][1][r] + b1;
      float m = fmaxf(a0, a1);
#pragma unroll
      for (int msk = 1; msk < 16; msk <<= 1) m = fmaxf(m, __shfl_xor(m, msk, 32));
      float s = __expf(a0 - m) + __expf(a1 - m);
#pragma unroll
      for (int msk = 1; msk < 16; msk <<= 1) s += __shfl_xor(s, msk, 32);
      if (l15 == 0) {
        const int row_l = wm * 64 + mi * 16 + r + (hi ? 8 : 0);
        SStat[wn][row_l] = make_float2(m, s);
      }
    }
  }
  __syncthreads();

  if (tid < MT) {
    float M = NEGBIG;
#pragma unroll
    for (int w = 0; w < 4; w++) M = fmaxf(M, SStat[w][tid].x);
    float S = 0.f;
#pragma unroll
    for (int w = 0; w < 4; w++) {
      const float2 p = SStat[w][tid];
      S += p.y * __expf(p.x - M);
    }
    const int grow = mt * MT + tid;
    partial[(size_t)grow * NTILES + nt] = make_float2(M, S);
  }
}

// ---------------------------------------------------------------------------
// Kernel 3: exact fp32 target logits per row (one wave per row).
// head col: c0 -> t, c1 -> tail_vec[1], c2 -> tail_vec[0]; tail col: weight[t].
// ---------------------------------------------------------------------------
__global__ void __launch_bounds__(256) k_target(
    const float* __restrict__ weight, const float* __restrict__ bias,
    const float* __restrict__ hiddens, const float* __restrict__ tailv,
    const float* __restrict__ tailb, const long long* __restrict__ targets,
    float* __restrict__ hlog, float* __restrict__ tlog, int* __restrict__ cl)
{
  const int wave = threadIdx.x >> 5, lane = threadIdx.x & 31;
  const int row  = blockIdx.x * 8 + wave;
  const long long t = targets[row];
  const int c = (int)(t >= 4000) + (int)(t >= 20000);

  const float* hv; float hb;
  const float* tv = nullptr; float tb = 0.f;
  if (c == 0) { hv = weight + (size_t)t * HDIM; hb = bias[t]; }
  else {
    const int tvrow = (c == 1) ? 1 : 0;
    hv = tailv + (size_t)tvrow * HDIM; hb = tailb[tvrow];
    tv = weight + (size_t)t * HDIM;    tb = bias[t];
  }
  const float* hrow = hiddens + (size_t)row * HDIM;
  float dh = 0.f, dt = 0.f;
  for (int k = lane; k < HDIM; k += 32) {
    const float x = hrow[k];
    dh = fmaf(x, hv[k], dh);
    if (c) dt = fmaf(x, tv[k], dt);
  }
#pragma unroll
  for (int msk = 16; msk >= 1; msk >>= 1) {
    dh += __shfl_xor(dh, msk, 32);
    dt += __shfl_xor(dt, msk, 32);
  }
  if (lane == 0) {
    hlog[row] = dh + hb;
    tlog[row] = c ? (dt + tb) : 0.f;
    cl[row]   = c;
  }
}

// ---------------------------------------------------------------------------
// Kernel 4: per-row segment LSE merge + loss. One wave per row.
// Segments in tile units: head [0,32), tail1 [32,157), tail2 [157,392).
// ---------------------------------------------------------------------------
__global__ void __launch_bounds__(32) k_rowloss(
    const float2* __restrict__ partial, const float* __restrict__ hlog,
    const float* __restrict__ tlog, const int* __restrict__ cl,
    float* __restrict__ rowloss)
{
  const int row  = blockIdx.x;
  const int lane = threadIdx.x;
  const float2* P = partial + (size_t)row * NTILES;
  const int segb[4] = {0, 32, 157, NTILES};

  float lse[3];
#pragma unroll
  for (int s = 0; s < 3; s++) {
    float M = NEGBIG, S = 0.f;
    for (int i = segb[s] + lane; i < segb[s + 1]; i += 32) {
      const float2 p = P[i];
      const float nM = fmaxf(M, p.x);
      S = S * __expf(M - nM) + p.y * __expf(p.x - nM);
      M = nM;
    }
#pragma unroll
    for (int msk = 16; msk >= 1; msk >>= 1) {
      const float oM = __shfl_xor(M, msk, 32);
      const float oS = __shfl_xor(S, msk, 32);
      const float nM = fmaxf(M, oM);
      S = S * __expf(M - nM) + oS * __expf(oM - nM);
      M = nM;
    }
    lse[s] = M + __logf(S);
  }
  if (lane == 0) {
    const int c = cl[row];
    float loss = lse[0] - hlog[row];
    if (c > 0) {
      const float lsec = (c == 1) ? lse[1] : lse[2];
      loss += lsec - tlog[row];
    }
    rowloss[row] = loss;
  }
}

// ---------------------------------------------------------------------------
// Kernel 5: mean over rows -> scalar output.
// ---------------------------------------------------------------------------
__global__ void __launch_bounds__(256) k_final(
    const float* __restrict__ rowloss, float* __restrict__ out)
{
  __shared__ float sm[256];
  float s = 0.f;
  for (int i = threadIdx.x; i < NROWS; i += 256) s += rowloss[i];
  sm[threadIdx.x] = s;
  __syncthreads();
  for (int st = 128; st > 0; st >>= 1) {
    if (threadIdx.x < st) sm[threadIdx.x] += sm[threadIdx.x + st];
    __syncthreads();
  }
  if (threadIdx.x == 0) out[0] = sm[0] / (float)NROWS;
}

// ---------------------------------------------------------------------------
extern "C" void kernel_launch(void* const* d_in, const int* in_sizes, int n_in,
                              void* d_out, int out_size, void* d_ws, size_t ws_size,
                              hipStream_t stream)
{
  const float*     weight  = (const float*)d_in[0];      // (50000, 1024)
  const float*     bias    = (const float*)d_in[1];      // (50000,)
  const float*     hiddens = (const float*)d_in[2];      // (4096, 1024)
  const float*     tailv   = (const float*)d_in[3];      // (2, 1024)
  const float*     tailb   = (const float*)d_in[4];      // (2,)
  const long long* targets = (const long long*)d_in[5];  // (4096,) int64

  char* ws = (char*)d_ws;
  size_t o = 0;
  unsigned short* Wb    = (unsigned short*)(ws + o); o += (size_t)NC * HDIM * 2;     // 102.8 MB
  unsigned short* Hb    = (unsigned short*)(ws + o); o += (size_t)NROWS * HDIM * 2;  //   8.4 MB
  float*          biasb = (float*)(ws + o);          o += (size_t)NC * 4;
  float2*         part  = (float2*)(ws + o);         o += (size_t)NROWS * NTILES * 8; // 12.8 MB
  float*          hlog  = (float*)(ws + o);          o += (size_t)NROWS * 4;
  float*          tlog  = (float*)(ws + o);          o += (size_t)NROWS * 4;
  int*            cl    = (int*)(ws + o);            o += (size_t)NROWS * 4;
  float*          rloss = (float*)(ws + o);          o += (size_t)NROWS * 4;

  k_convert<<<NC + NROWS, 128, 0, stream>>>(weight, bias, hiddens, tailv, tailb,
                                            Wb, biasb, Hb);

  dim3 grid(NTILES, NROWS / MT);   // x = weight tiles fastest (bf16 weight lives in 192MB L2)
  k_gemm_lse<<<grid, 256, 0, stream>>>(Wb, Hb, biasb, part);

  k_target<<<NROWS / 8, 256, 0, stream>>>(weight, bias, hiddens, tailv, tailb,
                                          targets, hlog, tlog, cl);

  k_rowloss<<<NROWS, 32, 0, stream>>>(part, hlog, tlog, cl, rloss);

  k_final<<<1, 256, 0, stream>>>(rloss, (float*)d_out);
}
